// BioSphericalCKN1D_87917980549725
// MI455X (gfx1250) — matrix operand order repaired and checked by reading, not compiled, not run on hardware
//
#include <hip/hip_runtime.h>

typedef __attribute__((ext_vector_type(16))) _Float16 v16h;
typedef __attribute__((ext_vector_type(8)))  float    v8f;
typedef __attribute__((ext_vector_type(4)))  unsigned int u32x4;
typedef __attribute__((ext_vector_type(4)))  int      i32x4;
typedef __attribute__((ext_vector_type(8)))  int      i32x8;

#define B_     16
#define L_     32768
#define C_     21
#define K_     9
#define F_     128
#define LOUT   32760
#define KC     189               // K_*C_ (GEMM reduction dim)
#define TILE_L 128
#define NLT    256               // ceil(LOUT/TILE_L)
#define XROWS  (TILE_L + K_ - 1) // 136
#define SLABN  (XROWS * C_)      // 2856 fp32 elements

__global__ __launch_bounds__(256)
void ckn1d_wmma_kernel(const float* __restrict__ x,
                       const float* __restrict__ kern,
                       const float* __restrict__ scale_p,
                       const float* __restrict__ bias,
                       float* __restrict__ out)
{
    // B in WMMA-fragment-native layout: [ntile][kchunk][lane][j]
    __shared__ _Float16 sB[8][6][32][16];          // 48 KB
    __shared__ __align__(16) float sSlab[SLABN];   // 11.2 KB raw fp32 slab (TDM dest)
    __shared__ float sE[XROWS];                    // per-row sum of squares
    __shared__ float sPE[TILE_L];                  // 9-tap sliding patch energy
    __shared__ float sBias[F_];

    const int tid  = threadIdx.x;
    const int lane = tid & 31;
    const int wv   = tid >> 5;        // wave id == M-tile index (0..7)
    const int lt   = blockIdx.x;      // L tile
    const int b    = blockIdx.y;      // batch
    const int l0   = lt * TILE_L;

    const int rows_avail = L_ - l0;                       // >= 128 always
    const int nel = (rows_avail < XROWS ? rows_avail : XROWS) * C_;

    // ---- phase 1a: wave 0 launches the TDM slab copy (global -> LDS, async) ----
#if __has_builtin(__builtin_amdgcn_tensor_load_to_lds)
    if (wv == 0) {
        const uint64_t ga  = (uint64_t)(uintptr_t)(x + ((size_t)b * L_ + l0) * C_);
        const uint32_t lds = (uint32_t)(uintptr_t)&sSlab[0];
        u32x4 g0;
        g0.x = 1u;                                    // count=1, load, no gather
        g0.y = lds;                                   // lds_addr
        g0.z = (uint32_t)ga;                          // global_addr[31:0]
        g0.w = (uint32_t)(ga >> 32) | 0x80000000u;    // global_addr hi | type=2
        i32x8 g1;
        g1[0] = 0x00020000;                           // data_size = 4B, no mask/pad
        g1[1] = (nel & 0xffff) << 16;                 // tensor_dim0[15:0]
        g1[2] = ((nel >> 16) & 0xffff) | (1 << 16);   // tensor_dim0 hi | tensor_dim1=1
        g1[3] = (nel & 0xffff) << 16;                 // tile_dim0
        g1[4] = 0;                                    // tile_dim1/2 unused
        g1[5] = nel;                                  // tensor_dim0_stride lo
        g1[6] = 0;
        g1[7] = 0;
        const i32x4 gz4 = {0, 0, 0, 0};
        const i32x8 gz8 = {0, 0, 0, 0, 0, 0, 0, 0};
        // 6-arg form (amdgpu-toolchain / clang-23): extra int32x8 group, then cpol
        __builtin_amdgcn_tensor_load_to_lds(g0, g1, gz4, gz4, gz8, 0);
        __builtin_amdgcn_s_wait_tensorcnt(0);
    }
#else
    for (int i = tid; i < nel; i += 256)
        sSlab[i] = x[((size_t)b * L_ + l0) * C_ + i];
#endif

    // ---- phase 1b (overlaps TDM): bias + kernel->B-fragment staging ----
    if (tid < F_) sBias[tid] = bias[tid];
    // fragment element j of lane ln covers K = kc*32 + (ln>>4)*16 + j,
    // column f = ntile*16 + (ln&15).  K padded 189 -> 192 with zeros.
    for (int idx = tid; idx < 8 * 6 * 32; idx += 256) {
        const int n  = idx / 192;
        const int r  = idx - n * 192;
        const int kc = r >> 5;
        const int ln = r & 31;
        const int f  = n * 16 + (ln & 15);
        const int kb = kc * 32 + ((ln >> 4) << 4);
        #pragma unroll
        for (int j = 0; j < 16; ++j) {
            const int kk = kb + j;
            sB[n][kc][ln][j] = (_Float16)((kk < KC) ? kern[kk * F_ + f] : 0.f);
        }
    }
    __syncthreads();   // slab + sB visible to all waves

    // ---- phase 2: exact fp32 row energies from the LDS slab ----
    if (tid < XROWS) {
        float ss = 0.f;
        if (l0 + tid < L_) {
            #pragma unroll
            for (int c = 0; c < C_; ++c) {
                const float v = sSlab[tid * C_ + c];
                ss += v * v;
            }
        }
        sE[tid] = ss;
    }
    __syncthreads();

    // ---- phase 3a: sliding 9-tap patch energies (threads 0..127) ----
    if (tid < TILE_L) {
        float pe = 0.f;
        #pragma unroll
        for (int t = 0; t < K_; ++t) pe += sE[tid + t];
        sPE[tid] = pe;
    }

    // ---- phase 3b (all waves, overlaps 3a): gather A fragments from slab ----
    // A 16x32 fragment, contiguous k order used here (same as sB): lane m = ln&15; element j covers
    // K = (ln>>4)*16 + j; im2col kk = tap*C_ + c.
    v16h afr[6];
    const int m     = lane & 15;
    const int khalf = (lane >> 4) << 4;   // 0 or 16
    #pragma unroll
    for (int kc = 0; kc < 6; ++kc) {
        v16h a;
        #pragma unroll
        for (int j = 0; j < 16; ++j) {
            const int kk = kc * 32 + khalf + j;
            _Float16 v = (_Float16)0.f;
            if (kk < KC) {
                const int tap = kk / C_;
                const int c   = kk - tap * C_;
                v = (_Float16)sSlab[(wv * 16 + m + tap) * C_ + c];
            }
            a[j] = v;
        }
        afr[kc] = a;
    }
    __syncthreads();   // sPE ready; (sB already ready)

    // ---- main WMMA loop: software-pipelined B-fragment loads ----
    v8f acc[8] = {};
    v16h bf = *(const v16h*)(&sB[0][0][lane][0]);
    #pragma unroll
    for (int kc = 0; kc < 6; ++kc) {
        #pragma unroll
        for (int n = 0; n < 8; ++n) {
            const v16h bcur = bf;
            if (!(kc == 5 && n == 7)) {          // prefetch next fragment
                const int n2  = (n + 1) & 7;
                const int kc2 = (n == 7) ? kc + 1 : kc;
                bf = *(const v16h*)(&sB[n2][kc2][lane][0]);
            }
            acc[n] = __builtin_amdgcn_wmma_f32_16x16x32_f16(
                false, afr[kc], false, bcur, (short)0, acc[n], false, false);
        }
    }

    // ---- epilogue: normalize, scale, bias, coalesced stores ----
    // C/D layout: VGPR r holds M = r + 8*(lane>>4), N = lane&15.
    const float scl = scale_p[0];
    const int   nl  = lane & 15;
    const int   hf  = lane >> 4;
    #pragma unroll
    for (int r = 0; r < 8; ++r) {
        const int   ml  = wv * 16 + r + hf * 8;
        const int   row = l0 + ml;
        const float pe  = sPE[ml];
        const float inv = scl / (sqrtf(pe + 1e-5f) + 1e-5f);
        if (row < LOUT) {
            float* po = out + ((size_t)b * LOUT + row) * F_;
            #pragma unroll
            for (int n = 0; n < 8; ++n) {
                const int f = n * 16 + nl;
                po[f] = acc[n][r] * inv + sBias[f];
            }
        }
    }
}

extern "C" void kernel_launch(void* const* d_in, const int* in_sizes, int n_in,
                              void* d_out, int out_size, void* d_ws, size_t ws_size,
                              hipStream_t stream) {
    (void)in_sizes; (void)n_in; (void)out_size; (void)d_ws; (void)ws_size;
    const float* x    = (const float*)d_in[0];
    const float* kern = (const float*)d_in[1];
    const float* scl  = (const float*)d_in[2];
    const float* bias = (const float*)d_in[3];
    float* out = (float*)d_out;
    dim3 grid(NLT, B_);
    ckn1d_wmma_kernel<<<grid, dim3(256), 0, stream>>>(x, kern, scl, bias, out);
}
